// RNN_78477642432863
// MI455X (gfx1250) — matrix operand order repaired and checked by reading, not compiled, not run on hardware
//
#include <hip/hip_runtime.h>
#include <math.h>

#define BB 128
#define TT 512
#define II 512
#define HH 1024
#define HD 64
#define OO 10

typedef __attribute__((ext_vector_type(16))) __bf16 v16bf;
typedef __attribute__((ext_vector_type(8)))  __bf16 v8bf;
typedef __attribute__((ext_vector_type(8)))  float  v8f;

__device__ __forceinline__ __bf16 f2bf(float f) {
  unsigned u = __builtin_bit_cast(unsigned, f);
  unsigned r = u + 0x7FFFu + ((u >> 16) & 1u);   // round-to-nearest-even
  unsigned short h = (unsigned short)(r >> 16);
  return __builtin_bit_cast(__bf16, h);
}
__device__ __forceinline__ float bf2f(__bf16 b) {
  unsigned short h = __builtin_bit_cast(unsigned short, b);
  unsigned u = ((unsigned)h) << 16;
  return __builtin_bit_cast(float, u);
}

// A-matrix 16x32 bf16 fragment (ISA 7.12.2): lanes 0-15 hold K={0..7,16..23},
// lanes 16-31 hold K={8..15,24..31} of row M=lane&15. Two 16B contiguous loads.
__device__ __forceinline__ v16bf ld_a_frag(const __bf16* __restrict__ row, int lane) {
  const int hi = (lane >> 4) << 3;               // 0 or 8
  v8bf lo = *(const v8bf*)(row + hi);
  v8bf hp = *(const v8bf*)(row + hi + 16);
  v16bf r;
#pragma unroll
  for (int j = 0; j < 8; ++j) { r[j] = lo[j]; r[j + 8] = hp[j]; }
  return r;
}

// B-matrix 32x16 bf16 fragment: lanes 0-15 hold K={0..7,16..23}, lanes 16-31 hold
// K={8..15,24..31} of column N=lane&15 (row of W since we compute X*W^T).
__device__ __forceinline__ v16bf ld_b_frag(const __bf16* __restrict__ row, int lane) {
  const int off = (lane >> 4) << 3;              // 0 or 8
  v8bf lo = *(const v8bf*)(row + off);
  v8bf hp = *(const v8bf*)(row + off + 16);
  v16bf r;
#pragma unroll
  for (int j = 0; j < 8; ++j) { r[j] = lo[j]; r[j + 8] = hp[j]; }
  return r;
}

__device__ __forceinline__ v8f wmma_bf16(v16bf a, v16bf b, v8f c) {
  return __builtin_amdgcn_wmma_f32_16x16x32_bf16(
      false, a, false, b, (short)0, c, false, false);
}

// ---------------- f32 -> bf16 conversion ----------------
__global__ __launch_bounds__(256) void f32_to_bf16_kernel(
    const float* __restrict__ in, __bf16* __restrict__ out, int n) {
  int i = blockIdx.x * blockDim.x + threadIdx.x;
  if (i < n) out[i] = f2bf(in[i]);
}

// ---------------- init: zero h0 + 8 per-group barrier counters ----------------
__global__ __launch_bounds__(256) void init_kernel(
    __bf16* __restrict__ h0, unsigned* __restrict__ counters, int n) {
  int i = blockIdx.x * blockDim.x + threadIdx.x;
  if (i < 8) counters[i] = 0u;
  if (i < n) h0[i] = __builtin_bit_cast(__bf16, (unsigned short)0);
}

// ---------------- XW = x @ W_x^T + bias  (bf16 WMMA, f32 out) ----------------
// 32x32 output block per wave: 2 A-frags x 2 B-frags -> 4 wmma per k-step.
// Rows r = t*BB + b read x[b,t,:]; output XW is [TT, BB, HH] f32 (bias folded).
__global__ __launch_bounds__(128) void xw_gemm_kernel(
    const __bf16* __restrict__ Xbf, const __bf16* __restrict__ Wxbf,
    const float* __restrict__ bias, float* __restrict__ XW) {
  const int lane = threadIdx.x & 31;
  const int wave = blockIdx.x * 4 + (threadIdx.x >> 5);
  const int nb = wave & 31;                      // HH/32 = 32 n-blocks
  const int mb = wave >> 5;                      // TT*BB/32 = 2048 m-blocks
  const int m0 = mb << 5, n0 = nb << 5;

  const int r0 = m0 + (lane & 15);
  const int r1 = r0 + 16;
  // r = t*BB + b  ->  t = r>>7, b = r&127 ; x is [BB, TT, II]
  const __bf16* a0p = Xbf + ((size_t)(r0 & 127) * TT + (r0 >> 7)) * II;
  const __bf16* a1p = Xbf + ((size_t)(r1 & 127) * TT + (r1 >> 7)) * II;
  const __bf16* b0p = Wxbf + (size_t)(n0 + (lane & 15)) * II;
  const __bf16* b1p = b0p + (size_t)16 * II;

  v8f acc00 = {}, acc01 = {}, acc10 = {}, acc11 = {};
#pragma unroll 2
  for (int k0 = 0; k0 < II; k0 += 32) {
    v16bf a0 = ld_a_frag(a0p + k0, lane);
    v16bf a1 = ld_a_frag(a1p + k0, lane);
    v16bf b0 = ld_b_frag(b0p + k0, lane);
    v16bf b1 = ld_b_frag(b1p + k0, lane);
    acc00 = wmma_bf16(a0, b0, acc00);
    acc01 = wmma_bf16(a0, b1, acc01);
    acc10 = wmma_bf16(a1, b0, acc10);
    acc11 = wmma_bf16(a1, b1, acc11);
  }

  const int c0 = n0 + (lane & 15), c1 = c0 + 16;
  const float bv0 = bias[c0], bv1 = bias[c1];
  const int rb0 = m0 + ((lane >> 4) << 3);       // C/D: M = v + 8*(lane>=16)
  const int rb1 = rb0 + 16;
#pragma unroll
  for (int v = 0; v < 8; ++v) {
    XW[(size_t)(rb0 + v) * HH + c0] = acc00[v] + bv0;
    XW[(size_t)(rb0 + v) * HH + c1] = acc01[v] + bv1;
    XW[(size_t)(rb1 + v) * HH + c0] = acc10[v] + bv0;
    XW[(size_t)(rb1 + v) * HH + c1] = acc11[v] + bv1;
  }
}

// ---------------- persistent tanh-RNN recurrence ----------------
// 128 blocks x 128 threads = 512 waves; wave owns one 16x16 tile of h.
// W_h fragments (32 x v16bf = 256 VGPRs) are loaded ONCE and stay register-
// resident for all 512 steps. Batch rows are independent, so barriers span
// only the 16 blocks (64 waves) of one 16-row m-group (8 separate counters).
__global__ __launch_bounds__(128, 1) void rnn_recurrence_kernel(
    const float* __restrict__ XWb, const __bf16* __restrict__ Whbf,
    __bf16* __restrict__ h0, __bf16* __restrict__ h1,
    unsigned* __restrict__ counters) {
  const int lane = threadIdx.x & 31;
  const int mgroup = blockIdx.x >> 4;                   // 0..7
  const int gidx   = blockIdx.x & 15;                   // 0..15
  const int ntile  = gidx * 4 + (threadIdx.x >> 5);     // 0..63
  const int m0 = mgroup << 4;
  const int n0 = ntile << 4;

  const int mrow = m0 + (lane & 15);
  const int ncol = n0 + (lane & 15);
  const int rowbase = m0 + ((lane >> 4) << 3);
  unsigned* cnt = counters + mgroup;

  // Preload this wave's W_h slice (16 rows x K=1024) into registers.
  const __bf16* wrow = Whbf + (size_t)ncol * HH;
  v16bf w[32];
#pragma unroll
  for (int kk = 0; kk < 32; ++kk) w[kk] = ld_b_frag(wrow + kk * 32, lane);

#pragma unroll 1
  for (int t = 0; t < TT; ++t) {
    const __bf16* Hin = (t & 1) ? h1 : h0;
    __bf16* Hout = (t & 1) ? h0 : h1;
    const float* xw_t = XWb + (size_t)t * (BB * HH);

    // Issue the (barrier-independent) XW loads first to overlap the chain.
    float xwv[8];
#pragma unroll
    for (int v = 0; v < 8; ++v)
      xwv[v] = xw_t[(size_t)(rowbase + v) * HH + ncol];

    const __bf16* arow = Hin + (size_t)mrow * HH;
    v8f acc = {};
#pragma unroll
    for (int kk = 0; kk < 32; ++kk) {
      v16bf a = ld_a_frag(arow + kk * 32, lane);
      acc = wmma_bf16(a, w[kk], acc);
    }

#pragma unroll
    for (int v = 0; v < 8; ++v)
      Hout[(size_t)(rowbase + v) * HH + ncol] = f2bf(tanhf(acc[v] + xwv[v]));

    // ---- m-group barrier (16 blocks, monotonic counter, rel/acq) ----
    __syncthreads();
    if (threadIdx.x == 0) {
      __threadfence();                                   // release stores
      atomicAdd(cnt, 1u);
      const unsigned target = (unsigned)(t + 1) * 16u;
      while (__hip_atomic_load(cnt, __ATOMIC_ACQUIRE,
                               __HIP_MEMORY_SCOPE_AGENT) < target) {
        __builtin_amdgcn_s_sleep(1);
      }
      __threadfence();                                   // acquire
    }
    __syncthreads();
  }
}

// ---------------- head: relu -> fc64 -> relu -> fc10 -> softmax ----------------
__global__ __launch_bounds__(64) void head_kernel(
    const __bf16* __restrict__ Hf, const float* __restrict__ W1,
    const float* __restrict__ b1, const float* __restrict__ W2,
    const float* __restrict__ b2, float* __restrict__ out) {
  const int b = blockIdx.x;
  const int j = threadIdx.x;
  __shared__ float hs[HH];
  __shared__ float z[HD];
  __shared__ float lg[16];

  for (int k = j; k < HH; k += HD) {
    float v = bf2f(Hf[(size_t)b * HH + k]);
    hs[k] = v > 0.f ? v : 0.f;
  }
  __syncthreads();

  float acc = b1[j];
  const float* w = W1 + (size_t)j * HH;
  for (int k = 0; k < HH; ++k) acc = fmaf(hs[k], w[k], acc);
  z[j] = acc > 0.f ? acc : 0.f;
  __syncthreads();

  if (j < OO) {
    float a = b2[j];
    const float* w2 = W2 + j * HD;
#pragma unroll
    for (int k = 0; k < HD; ++k) a = fmaf(z[k], w2[k], a);
    lg[j] = a;
  }
  __syncthreads();

  if (j < OO) {
    float mx = lg[0];
    for (int k = 1; k < OO; ++k) mx = fmaxf(mx, lg[k]);
    float s = 0.f;
    for (int k = 0; k < OO; ++k) s += __expf(lg[k] - mx);
    out[b * OO + j] = __expf(lg[j] - mx) / s;
  }
}

extern "C" void kernel_launch(void* const* d_in, const int* in_sizes, int n_in,
                              void* d_out, int out_size, void* d_ws, size_t ws_size,
                              hipStream_t stream) {
  (void)in_sizes; (void)n_in; (void)out_size; (void)ws_size;
  const float* x   = (const float*)d_in[0];   // [BB, TT, II]
  const float* W_h = (const float*)d_in[1];   // [HH, HH]
  const float* W_x = (const float*)d_in[2];   // [HH, II]
  const float* bia = (const float*)d_in[3];   // [HH]
  const float* W1  = (const float*)d_in[4];   // [64, HH]
  const float* b1  = (const float*)d_in[5];   // [64]
  const float* W2  = (const float*)d_in[6];   // [10, 64]
  const float* b2  = (const float*)d_in[7];   // [10]
  float* out = (float*)d_out;

  char* ws = (char*)d_ws;
  size_t oXbf = 0;
  size_t oWx  = oXbf + (size_t)BB * TT * II * 2;   //  64 MB
  size_t oWh  = oWx  + (size_t)HH * II * 2;        // + 1 MB
  size_t oXW  = oWh  + (size_t)HH * HH * 2;        // + 2 MB
  size_t oH0  = oXW  + (size_t)TT * BB * HH * 4;   // +256 MB
  size_t oH1  = oH0  + (size_t)BB * HH * 2;
  size_t oCnt = oH1  + (size_t)BB * HH * 2;

  __bf16*   Xbf  = (__bf16*)(ws + oXbf);
  __bf16*   Wxbf = (__bf16*)(ws + oWx);
  __bf16*   Whbf = (__bf16*)(ws + oWh);
  float*    XW   = (float*)(ws + oXW);
  __bf16*   h0   = (__bf16*)(ws + oH0);
  __bf16*   h1   = (__bf16*)(ws + oH1);
  unsigned* cnt  = (unsigned*)(ws + oCnt);

  // 1) convert inputs/weights to bf16
  {
    int n = BB * TT * II;
    f32_to_bf16_kernel<<<(n + 255) / 256, 256, 0, stream>>>(x, Xbf, n);
  }
  {
    int n = HH * II;
    f32_to_bf16_kernel<<<(n + 255) / 256, 256, 0, stream>>>(W_x, Wxbf, n);
  }
  {
    int n = HH * HH;
    f32_to_bf16_kernel<<<(n + 255) / 256, 256, 0, stream>>>(W_h, Whbf, n);
  }

  // 2) zero h0 + barrier counters (ws is poisoned; must re-init every call)
  {
    int n = BB * HH;
    init_kernel<<<(n + 255) / 256, 256, 0, stream>>>(h0, cnt, n);
  }

  // 3) XW = x @ W_x^T + bias for all timesteps (32x32-blocked WMMA GEMM)
  {
    int waves = (TT * BB / 32) * (HH / 32);      // 65536 waves
    xw_gemm_kernel<<<waves / 4, 128, 0, stream>>>(Xbf, Wxbf, bia, XW);
  }

  // 4) persistent recurrence: 128 blocks x 4 waves = 512 tiles of h
  rnn_recurrence_kernel<<<128, 128, 0, stream>>>(XW, Whbf, h0, h1, cnt);

  // 5) MLP head + softmax (final h is in h0 since TT is even)
  head_kernel<<<BB, HD, 0, stream>>>(h0, W1, b1, W2, b2, out);
}
